// RecurrentGemmaRglru_85366769975584
// MI455X (gfx1250) — matrix operand order repaired and checked by reading, not compiled, not run on hardware
//
#include <hip/hip_runtime.h>

typedef __bf16 v16bf __attribute__((ext_vector_type(16)));
typedef float  v8f   __attribute__((ext_vector_type(8)));

#define B_  4
#define T_  4096
#define D_  2560
#define H_  10
#define BW_ 256
#define TC_ 64
#define NC_ (T_/TC_)

// padded row stride (in u32 pairs) for packed bf16 X tiles: 133 % 64 == 5
// -> conflict-free LDS fragment gathers (see analysis)
#define PST_ 133

union FragU { v16bf bf; unsigned u[8]; };
union PairU { __bf16 b[2]; unsigned u; };

// -------------------------------------------------------------------------
// Prep: convert both weight matrices (f32) into k-pair-interleaved packed
// bf16 hi/lo layout:  Wp[(h*4 + gate*2 + part)*32768 + k2*256 + col]
// where u32 = { bf16(W[2*k2][col]) , bf16(W[2*k2+1][col]) }.
// -------------------------------------------------------------------------
__global__ __launch_bounds__(256) void rglru_prep(
    const float* __restrict__ Wig, const float* __restrict__ Wrg,
    unsigned* __restrict__ Wp)
{
    const int g = blockIdx.x * 256 + threadIdx.x;   // H_*2*128*256 = 655360
    if (g >= H_ * 2 * 128 * 256) return;
    const int col  = g & 255;
    const int k2   = (g >> 8) & 127;
    const int gate = (g >> 15) & 1;
    const int h    = g >> 16;
    const float* W = (gate ? Wrg : Wig) + (size_t)h * BW_ * BW_;
    const float w0 = W[(size_t)(2 * k2) * BW_ + col];
    const float w1 = W[(size_t)(2 * k2 + 1) * BW_ + col];
    const __bf16 h0 = (__bf16)w0, h1 = (__bf16)w1;
    PairU ph, pl;
    ph.b[0] = h0;
    ph.b[1] = h1;
    pl.b[0] = (__bf16)(w0 - (float)h0);
    pl.b[1] = (__bf16)(w1 - (float)h1);
    const size_t base = (size_t)(h * 4 + gate * 2) * 32768 + (size_t)k2 * 256 + col;
    Wp[base]         = ph.u;
    Wp[base + 32768] = pl.u;
}

// -------------------------------------------------------------------------
// Phase 1: per (b, head, time-chunk) block:
//   gates via WMMA bf16 hi/lo GEMM, elementwise RG-LRU, local scan.
// -------------------------------------------------------------------------
__global__ __launch_bounds__(256) void rglru_phase1(
    const float* __restrict__ X,  const int* __restrict__ Pos,
    const unsigned* __restrict__ Wp,
    const float* __restrict__ Big, const float* __restrict__ Brg,
    const float* __restrict__ RP,
    float* __restrict__ Y, float* __restrict__ Cum)
{
    __shared__ float    sMain[TC_ * BW_];     // X (f32), later x_norm for scan
    __shared__ unsigned sAux[2 * TC_ * PST_]; // packed bf16 X hi/lo, later a_eff (f32)
    __shared__ int      sPos[TC_];

    unsigned* sXh = sAux;
    unsigned* sXl = sAux + TC_ * PST_;

    const int tid = threadIdx.x;
    const int bid = blockIdx.x;
    const int ck  = bid % NC_;
    const int hh  = (bid / NC_) % H_;
    const int bb  = bid / (NC_ * H_);
    const int t0  = ck * TC_;

    // ---- load X tile [TC_ x BW_], split to packed bf16 hi/lo once ----
    {
        const size_t gbase = ((size_t)bb * T_ + t0) * D_ + (size_t)hh * BW_;
        const float4* __restrict__ X4 = reinterpret_cast<const float4*>(X + gbase);
        float4* s4 = reinterpret_cast<float4*>(sMain);
        #pragma unroll
        for (int it = 0; it < 16; ++it) {
            const int lin = it * 256 + tid;   // 4096 float4 elements
            const int row = lin >> 6;
            const int c4  = lin & 63;
            const float4 v = X4[(size_t)row * (D_ / 4) + c4];
            s4[row * 64 + c4] = v;
            const __bf16 h0 = (__bf16)v.x, h1 = (__bf16)v.y;
            const __bf16 h2 = (__bf16)v.z, h3 = (__bf16)v.w;
            PairU p0, p1, q0, q1;
            p0.b[0] = h0; p0.b[1] = h1;
            p1.b[0] = h2; p1.b[1] = h3;
            q0.b[0] = (__bf16)(v.x - (float)h0);
            q0.b[1] = (__bf16)(v.y - (float)h1);
            q1.b[0] = (__bf16)(v.z - (float)h2);
            q1.b[1] = (__bf16)(v.w - (float)h3);
            sXh[row * PST_ + c4 * 2]     = p0.u;
            sXh[row * PST_ + c4 * 2 + 1] = p1.u;
            sXl[row * PST_ + c4 * 2]     = q0.u;
            sXl[row * PST_ + c4 * 2 + 1] = q1.u;
        }
        if (tid < TC_) sPos[tid] = Pos[bb * T_ + t0 + tid];
    }
    __syncthreads();

    const int wid   = tid >> 5;      // wave 0..7 -> owns n columns [32w, 32w+32)
    const int lane  = tid & 31;
    const int lan16 = lane & 15;
    const int half  = lane >> 4;

    v8f accI[4][2], accR[4][2];
    #pragma unroll
    for (int mt = 0; mt < 4; ++mt)
        #pragma unroll
        for (int nt = 0; nt < 2; ++nt) { v8f z = {}; accI[mt][nt] = z; accR[mt][nt] = z; }

    #pragma unroll
    for (int gate = 0; gate < 2; ++gate) {
        const unsigned* __restrict__ WH = Wp + (size_t)(hh * 4 + gate * 2) * 32768;
        const unsigned* __restrict__ WL = WH + 32768;
        v8f (*acc)[2] = gate ? accR : accI;

        for (int ks = 0; ks < 8; ++ks) {               // K in steps of 32
            // ---- B fragments (32x16 bf16 layout): direct packed loads from L2 ----
            FragU bh[2], bl[2];
            #pragma unroll
            for (int nt = 0; nt < 2; ++nt) {
                const int col = ((wid << 1) + nt) * 16 + lan16;
                #pragma unroll
                for (int v = 0; v < 8; ++v) {
                    const int k2 = (ks << 4) + (half << 3) + v;   // k pair index
                    bh[nt].u[v] = WH[(size_t)k2 * 256 + col];
                    bl[nt].u[v] = WL[(size_t)k2 * 256 + col];
                }
            }
            #pragma unroll
            for (int mt = 0; mt < 4; ++mt) {
                // ---- A fragments (16x32 bf16 layout): packed LDS gathers ----
                const int row = (mt << 4) + lan16;
                FragU ah, al;
                #pragma unroll
                for (int v = 0; v < 8; ++v) {
                    const int k2 = (ks << 4) + (half << 3) + v;
                    ah.u[v] = sXh[row * PST_ + k2];
                    al.u[v] = sXl[row * PST_ + k2];
                }
                #pragma unroll
                for (int nt = 0; nt < 2; ++nt) {
                    v8f c = acc[mt][nt];
                    c = __builtin_amdgcn_wmma_f32_16x16x32_bf16(false, ah.bf, false, bh[nt].bf, (short)0, c, false, false);
                    c = __builtin_amdgcn_wmma_f32_16x16x32_bf16(false, al.bf, false, bh[nt].bf, (short)0, c, false, false);
                    c = __builtin_amdgcn_wmma_f32_16x16x32_bf16(false, ah.bf, false, bl[nt].bf, (short)0, c, false, false);
                    acc[mt][nt] = c;
                }
            }
        }
    }

    // ---- elementwise RG-LRU math in C-matrix register layout ----
    #pragma unroll
    for (int nt = 0; nt < 2; ++nt) {
        const int col = ((wid << 1) + nt) * 16 + lan16;
        const int d   = hh * BW_ + col;
        const float bi = Big[d];
        const float br = Brg[d];
        const float r  = RP[d];
        const float sp = fmaxf(r, 0.f) + log1pf(expf(-fabsf(r)));  // softplus
        #pragma unroll
        for (int mt = 0; mt < 4; ++mt) {
            #pragma unroll
            for (int v = 0; v < 8; ++v) {
                const int tl = (mt << 4) + (half << 3) + v;        // local time row
                const float ig = 1.f / (1.f + expf(-(accI[mt][nt][v] + bi)));
                const float rg = 1.f / (1.f + expf(-(accR[mt][nt][v] + br)));
                const float la = -8.f * rg * sp;
                const float a  = expf(la);
                const bool reset = (sPos[tl] == 0);
                const float mult = reset ? 1.f : sqrtf(fmaxf(1.f - a * a, 0.f));
                const float x  = sMain[tl * BW_ + col];
                accI[mt][nt][v] = x * ig * mult;                   // x_norm
                accR[mt][nt][v] = reset ? 0.f : a;                 // a_eff
            }
        }
    }
    __syncthreads();

    float* sAf = reinterpret_cast<float*>(sAux);     // a_eff staging (overlays packed X)
    #pragma unroll
    for (int nt = 0; nt < 2; ++nt) {
        const int col = ((wid << 1) + nt) * 16 + lan16;
        #pragma unroll
        for (int mt = 0; mt < 4; ++mt)
            #pragma unroll
            for (int v = 0; v < 8; ++v) {
                const int tl = (mt << 4) + (half << 3) + v;
                sMain[tl * BW_ + col] = accI[mt][nt][v];
                sAf[tl * BW_ + col]   = accR[mt][nt][v];
            }
    }
    __syncthreads();

    // ---- local scan: one channel per thread, 64 steps (wave32 conflict-free) ----
    {
        float hcur = 0.f, cA = 1.f;
        const size_t base = ((size_t)bb * T_ + t0) * D_ + (size_t)hh * BW_ + tid;
        #pragma unroll 4
        for (int t = 0; t < TC_; ++t) {
            const float a = sAf[t * BW_ + tid];
            const float x = sMain[t * BW_ + tid];
            hcur = fmaf(a, hcur, x);
            cA *= a;
            Y[base + (size_t)t * D_]   = hcur;   // local-scan output (zero init state)
            Cum[base + (size_t)t * D_] = cA;     // cumulative decay within chunk
        }
    }
}

// -------------------------------------------------------------------------
// Phase 2: serial combine of chunk carries per (b, d) channel.
// -------------------------------------------------------------------------
__global__ __launch_bounds__(256) void rglru_phase2(
    const float* __restrict__ Y, const float* __restrict__ Cum, float* __restrict__ HIn)
{
    const int g = blockIdx.x * 256 + threadIdx.x;
    if (g >= B_ * D_) return;
    const int b = g / D_, d = g % D_;
    float hcur = 0.f;
    for (int c = 0; c < NC_; ++c) {
        HIn[((size_t)b * NC_ + c) * D_ + d] = hcur;
        const size_t idx = ((size_t)b * T_ + (size_t)c * TC_ + (TC_ - 1)) * D_ + d;
        hcur = fmaf(Cum[idx], hcur, Y[idx]);
    }
}

// -------------------------------------------------------------------------
// Phase 3: y[t] = y_local[t] + cumA[t] * h_in(chunk(t)), float4 vectorized.
// -------------------------------------------------------------------------
__global__ __launch_bounds__(256) void rglru_phase3(
    float* __restrict__ Y, const float* __restrict__ Cum, const float* __restrict__ HIn)
{
    const size_t g = (size_t)blockIdx.x * 256 + threadIdx.x;
    const size_t total4 = (size_t)B_ * T_ * D_ / 4;
    if (g >= total4) return;
    const size_t e = g * 4;
    const int    b = (int)(e / ((size_t)T_ * D_));
    const size_t r = e % ((size_t)T_ * D_);
    const int    t = (int)(r / D_);
    const int    d = (int)(r % D_);
    const int    c = t / TC_;
    if (c == 0) return;                      // h_in == 0 for the first chunk
    float4 y        = *reinterpret_cast<const float4*>(Y + e);
    const float4 ca = *reinterpret_cast<const float4*>(Cum + e);
    const float4 hi = *reinterpret_cast<const float4*>(HIn + ((size_t)b * NC_ + c) * D_ + d);
    y.x = fmaf(ca.x, hi.x, y.x);
    y.y = fmaf(ca.y, hi.y, y.y);
    y.z = fmaf(ca.z, hi.z, y.z);
    y.w = fmaf(ca.w, hi.w, y.w);
    *reinterpret_cast<float4*>(Y + e) = y;
}

extern "C" void kernel_launch(void* const* d_in, const int* in_sizes, int n_in,
                              void* d_out, int out_size, void* d_ws, size_t ws_size,
                              hipStream_t stream)
{
    (void)in_sizes; (void)n_in; (void)out_size; (void)ws_size;
    const float* X   = (const float*)d_in[0];
    const int*   Pos = (const int*)  d_in[1];
    const float* Wig = (const float*)d_in[2];
    const float* Big = (const float*)d_in[3];
    const float* Wrg = (const float*)d_in[4];
    const float* Brg = (const float*)d_in[5];
    const float* RP  = (const float*)d_in[6];
    float* Y = (float*)d_out;

    float*    Cum = (float*)d_ws;                        // B*T*D floats (167.8 MB)
    float*    HIn = Cum + (size_t)B_ * T_ * D_;          // B*NC*D floats (2.6 MB)
    unsigned* Wp  = (unsigned*)(HIn + (size_t)B_ * NC_ * D_);  // 1310720 u32 (5.2 MB)

    rglru_prep<<<(H_ * 2 * 128 * 256) / 256, 256, 0, stream>>>(Wig, Wrg, Wp);
    rglru_phase1<<<B_ * H_ * NC_, 256, 0, stream>>>(X, Pos, Wp, Big, Brg, RP, Y, Cum);
    rglru_phase2<<<(B_ * D_ + 255) / 256, 256, 0, stream>>>(Y, Cum, HIn);
    const size_t total4 = (size_t)B_ * T_ * D_ / 4;
    rglru_phase3<<<(unsigned)((total4 + 255) / 256), 256, 0, stream>>>(Y, Cum, HIn);
}
